// SelfAttention_37349035606587
// MI455X (gfx1250) — matrix operand order repaired and checked by reading, not compiled, not run on hardware
//
#include <hip/hip_runtime.h>

#define EMB 768
#define SEQ 2048
#define BATCH 8

typedef __attribute__((ext_vector_type(16))) __bf16 bf16x16;
typedef __attribute__((ext_vector_type(8)))  float  f32x8;
typedef __attribute__((ext_vector_type(16))) unsigned short u16x16;

struct B32x2 { uint4 lo, hi; };   // 32 bytes == one bf16x16 operand

// round-to-nearest-even f32 -> bf16
static __device__ inline unsigned short f2bf(float f) {
  unsigned int u = __float_as_uint(f);
  u += 0x7FFFu + ((u >> 16) & 1u);
  return (unsigned short)(u >> 16);
}

static __device__ inline unsigned int pk2(unsigned short lo, unsigned short hi) {
  return (unsigned int)lo | ((unsigned int)hi << 16);
}

// assemble a WMMA operand from two 16-byte chunks (global or LDS)
static __device__ inline bf16x16 load_op(const unsigned short* p0, const unsigned short* p1) {
  B32x2 u;
  u.lo = *(const uint4*)p0;
  u.hi = *(const uint4*)p1;
  return __builtin_bit_cast(bf16x16, u);
}

// ---------------------------------------------------------------------------
// Kernel 0: bulk fp32 -> bf16 conversion (bandwidth-bound).
// ---------------------------------------------------------------------------
__global__ __launch_bounds__(256) void cvt_kernel(const float* __restrict__ src,
                                                  unsigned short* __restrict__ dst,
                                                  int n8) {
  const int i = blockIdx.x * 256 + threadIdx.x;
  if (i >= n8) return;
  const float4 a = ((const float4*)src)[2 * i + 0];
  const float4 b = ((const float4*)src)[2 * i + 1];
  uint4 o;
  o.x = pk2(f2bf(a.x), f2bf(a.y));
  o.y = pk2(f2bf(a.z), f2bf(a.w));
  o.z = pk2(f2bf(b.x), f2bf(b.y));
  o.w = pk2(f2bf(b.z), f2bf(b.w));
  ((uint4*)dst)[i] = o;
}

// ---------------------------------------------------------------------------
// Kernel 1: shared linear projection  Y = X @ W^T + b, all-bf16 operands.
//   WHICH==0 : qp  (scaled by 1/sqrt(EMB)), row-major [B*S, EMB]
//   WHICH==1 : kp  row-major [B*S, EMB]
//   WHICH==2 : vpT transposed [B, EMB, S]
// ---------------------------------------------------------------------------
template <int WHICH>
__global__ __launch_bounds__(256) void proj_kernel(
    const unsigned short* __restrict__ X, const unsigned short* __restrict__ Wbf,
    const float* __restrict__ bias, unsigned short* __restrict__ dst)
{
  const int lane = threadIdx.x;         // 0..31
  const int l16  = lane & 15;
  const int hi   = lane >> 4;           // lane half selects K sub-block
  const int mtile = blockIdx.x;                       // 0..1023
  const int ntile = blockIdx.y * 8 + threadIdx.y;     // 0..47

  const int arow = mtile * 16 + l16;    // A: lane = M row
  const int ncol = ntile * 16 + l16;    // B/C: lane = N col

  const unsigned short* abase = X + (size_t)arow * EMB + hi * 16;
  const unsigned short* bbase = Wbf + (size_t)ncol * EMB + hi * 16;

  f32x8 acc = {};
#pragma unroll 4
  for (int kc = 0; kc < EMB / 32; ++kc) {
    const bf16x16 A = load_op(abase + kc * 32, abase + kc * 32 + 8);
    const bf16x16 B = load_op(bbase + kc * 32, bbase + kc * 32 + 8);
    acc = __builtin_amdgcn_wmma_f32_16x16x32_bf16(false, A, false, B,
                                                  (short)0, acc, false, false);
  }

  const float bias_v = bias[ncol];
  const float qscale = 0.03608439182435161f;   // 1/sqrt(768), folded into qp
#pragma unroll
  for (int r = 0; r < 8; ++r) {
    const int row = mtile * 16 + r + hi * 8;   // C layout: reg r -> row r + 8*hi
    const float y = acc[r] + bias_v;
    if constexpr (WHICH == 0) {
      dst[(size_t)row * EMB + ncol] = f2bf(y * qscale);
    } else if constexpr (WHICH == 1) {
      dst[(size_t)row * EMB + ncol] = f2bf(y);
    } else {
      const int bb = row >> 11;                // row / SEQ
      const int ss = row & (SEQ - 1);
      dst[((size_t)(bb * EMB + ncol)) * SEQ + ss] = f2bf(y);
    }
  }
}

// ---------------------------------------------------------------------------
// Kernel 2: flash attention. 8 waves per block, one (batch, 16-row q block).
// Wave w owns K-slice / D-slice [96w, 96w+96). Key blocks of 32.
// K/V operands are preloaded in bulk so the WMMA chain gets staggered
// s_wait_loadcnt instead of full serialization.
// ---------------------------------------------------------------------------
__global__ __launch_bounds__(256) void attn_kernel(
    const unsigned short* __restrict__ qp, const unsigned short* __restrict__ kp,
    const unsigned short* __restrict__ vpT, const int* __restrict__ mask,
    float* __restrict__ out)
{
  __shared__ float          Sbuf[16 * 32];   // score tile, f32
  __shared__ unsigned short Pbuf[16 * 32];   // probability tile, bf16

  const int tid  = threadIdx.x;
  const int lane = tid & 31;
  const int wid  = tid >> 5;        // 0..7
  const int l16  = lane & 15;
  const int hi   = lane >> 4;
  const int qt   = blockIdx.x;      // 0..127
  const int b    = blockIdx.y;      // 0..7
  const int kbase = wid * 96;       // this wave's K/D slice

  // resident Q A-tiles (16 x 96 bf16, A layout, loaded once)
  bf16x16 qA[3];
  {
    const unsigned short* qr =
        qp + ((size_t)(b * SEQ + qt * 16 + l16)) * EMB + kbase + hi * 16;
#pragma unroll
    for (int kk = 0; kk < 3; ++kk)
      qA[kk] = load_op(qr + kk * 32, qr + kk * 32 + 8);
  }

  const unsigned short* krow = kp + ((size_t)(b * SEQ + l16)) * EMB + kbase + hi * 16;
  const unsigned short* vrow = vpT + ((size_t)(b * EMB + kbase + l16)) * SEQ + hi * 16;

  f32x8 O[6] = {};                  // 16 x 96 f32 accumulator (6 N-tiles)
  float m_r[8], l_r[8];
#pragma unroll
  for (int r = 0; r < 8; ++r) { m_r[r] = -__builtin_inff(); l_r[r] = 0.0f; }

  for (int jb = 0; jb < SEQ; jb += 32) {
    // ---- bulk-preload all six K B-operands for this key block ----
    bf16x16 KB0[3], KB1[3];
#pragma unroll
    for (int kk = 0; kk < 3; ++kk) {
      const unsigned short* kr = krow + (size_t)jb * EMB + kk * 32;
      KB0[kk] = load_op(kr, kr + 8);
      KB1[kk] = load_op(kr + 16 * EMB, kr + 16 * EMB + 8);
    }

    // ---- zero score buffer ----
    Sbuf[tid] = 0.0f;
    Sbuf[tid + 256] = 0.0f;
    __syncthreads();

    // prefetch next key block's K and V rows into near cache
    if (jb + 32 < SEQ) {
      __builtin_prefetch(krow + (size_t)(jb + 32) * EMB, 0, 3);
      __builtin_prefetch(krow + (size_t)(jb + 48) * EMB, 0, 3);
      __builtin_prefetch(vrow + jb + 32, 0, 3);
    }

    // ---- partial scores: S = Q(slice) * K(slice)^T, two 16x16 key halves ----
    f32x8 S0 = {}, S1 = {};
#pragma unroll
    for (int kk = 0; kk < 3; ++kk)
      S0 = __builtin_amdgcn_wmma_f32_16x16x32_bf16(false, qA[kk], false, KB0[kk],
                                                   (short)0, S0, false, false);
#pragma unroll
    for (int kk = 0; kk < 3; ++kk)
      S1 = __builtin_amdgcn_wmma_f32_16x16x32_bf16(false, qA[kk], false, KB1[kk],
                                                   (short)0, S1, false, false);

#pragma unroll
    for (int r = 0; r < 8; ++r) {
      atomicAdd(&Sbuf[(r + hi * 8) * 32 + l16],      S0[r]);   // ds_add_f32
      atomicAdd(&Sbuf[(r + hi * 8) * 32 + l16 + 16], S1[r]);
    }
    __syncthreads();

    // ---- apply mask once, cooperatively (scale already folded into qp) ----
    {
      const int e = tid * 2;
      const int r = e >> 5, c = e & 31;
      const int* mp = mask + ((size_t)(b * SEQ + qt * 16 + r)) * SEQ + jb + c;
      const int2 mv = *(const int2*)mp;
      if (mv.x == 1) Sbuf[e]     = -__builtin_inff();
      if (mv.y == 1) Sbuf[e + 1] = -__builtin_inff();
    }
    __syncthreads();

    // ---- bulk-preload all six V B-operands for this key block ----
    bf16x16 VB[6];
#pragma unroll
    for (int t = 0; t < 6; ++t) {
      const unsigned short* vr = vrow + (size_t)(t * 16) * SEQ + jb;
      VB[t] = load_op(vr, vr + 8);
    }

    // ---- online softmax (all waves redundantly; co-executes with XDL) ----
    float p0[8], p1[8];
#pragma unroll
    for (int r = 0; r < 8; ++r) {
      const float s0 = Sbuf[(r + hi * 8) * 32 + l16];
      const float s1 = Sbuf[(r + hi * 8) * 32 + l16 + 16];
      float mx = fmaxf(s0, s1);
#pragma unroll
      for (int d = 1; d < 16; d <<= 1) mx = fmaxf(mx, __shfl_xor(mx, d, 32));
      const float mnew  = fmaxf(m_r[r], mx);
      const float alpha = __expf(m_r[r] - mnew);
      p0[r] = __expf(s0 - mnew);
      p1[r] = __expf(s1 - mnew);
      float rs = p0[r] + p1[r];
#pragma unroll
      for (int d = 1; d < 16; d <<= 1) rs += __shfl_xor(rs, d, 32);
      l_r[r] = l_r[r] * alpha + rs;
      m_r[r] = mnew;
#pragma unroll
      for (int t = 0; t < 6; ++t) O[t][r] *= alpha;
    }

    // ---- broadcast P through LDS (identical in all waves; wave 0 writes) ----
    if (wid == 0) {
#pragma unroll
      for (int r = 0; r < 8; ++r) {
        Pbuf[(r + hi * 8) * 32 + l16]      = f2bf(p0[r]);
        Pbuf[(r + hi * 8) * 32 + l16 + 16] = f2bf(p1[r]);
      }
    }
    __syncthreads();

    // ---- read P in A layout (two ds_load_b128 per lane) ----
    const unsigned short* pp = &Pbuf[l16 * 32 + hi * 16];
    const bf16x16 PA = load_op(pp, pp + 8);

    // ---- O += P * V for this wave's 96-wide D slice ----
#pragma unroll
    for (int t = 0; t < 6; ++t)
      O[t] = __builtin_amdgcn_wmma_f32_16x16x32_bf16(false, PA, false, VB[t],
                                                     (short)0, O[t], false, false);
  }

  // ---- normalize by softmax denominator and store ----
#pragma unroll
  for (int t = 0; t < 6; ++t) {
    const int d = kbase + t * 16 + l16;
#pragma unroll
    for (int r = 0; r < 8; ++r) {
      const int row = qt * 16 + r + hi * 8;
      out[((size_t)(b * SEQ + row)) * EMB + d] = O[t][r] / l_r[r];
    }
  }
}

// ---------------------------------------------------------------------------
extern "C" void kernel_launch(void* const* d_in, const int* in_sizes, int n_in,
                              void* d_out, int out_size, void* d_ws, size_t ws_size,
                              hipStream_t stream) {
  (void)in_sizes; (void)n_in; (void)out_size; (void)ws_size;
  const float* q    = (const float*)d_in[0];
  const float* k    = (const float*)d_in[1];
  const float* v    = (const float*)d_in[2];
  const int*   mask = (const int*)d_in[3];
  const float* W    = (const float*)d_in[4];
  const float* bias = (const float*)d_in[5];

  const size_t n  = (size_t)BATCH * SEQ * EMB;   // 12.58M elements
  const size_t nw = (size_t)EMB * EMB;           // 0.59M elements
  unsigned short* xq  = (unsigned short*)d_ws;
  unsigned short* xk  = xq + n;
  unsigned short* xv  = xk + n;
  unsigned short* Wbf = xv + n;
  unsigned short* qp  = Wbf + nw;
  unsigned short* kp  = qp + n;
  unsigned short* vpT = kp + n;

  // 0) bulk fp32 -> bf16 conversion (once per tensor, bandwidth-bound)
  const int n8  = (int)(n / 8);
  const int nw8 = (int)(nw / 8);
  cvt_kernel<<<(n8 + 255) / 256, 256, 0, stream>>>(q, xq, n8);
  cvt_kernel<<<(n8 + 255) / 256, 256, 0, stream>>>(k, xk, n8);
  cvt_kernel<<<(n8 + 255) / 256, 256, 0, stream>>>(v, xv, n8);
  cvt_kernel<<<(nw8 + 255) / 256, 256, 0, stream>>>(W, Wbf, nw8);

  // 1) projections (specialized epilogues, no runtime branching)
  dim3 pblk(32, 8);
  dim3 pgrid((BATCH * SEQ) / 16, (EMB / 16) / 8);
  proj_kernel<0><<<pgrid, pblk, 0, stream>>>(xq, Wbf, bias, qp);
  proj_kernel<1><<<pgrid, pblk, 0, stream>>>(xk, Wbf, bias, kp);
  proj_kernel<2><<<pgrid, pblk, 0, stream>>>(xv, Wbf, bias, vpT);

  // 2) flash attention
  dim3 ablk(256);
  dim3 agrid(SEQ / 16, BATCH);
  attn_kernel<<<agrid, ablk, 0, stream>>>(qp, kp, vpT, mask, (float*)d_out);
}
